// CARAFE_1288490189338
// MI455X (gfx1250) — compile-verified, matrix operand order checked by reading, not run on hardware
//
#include <hip/hip_runtime.h>
#include <hip/hip_bf16.h>
#include <math.h>

typedef __attribute__((ext_vector_type(16))) _Float16 v16h;
typedef __attribute__((ext_vector_type(8)))  _Float16 v8h;
typedef __attribute__((ext_vector_type(8)))  float    v8f;

#define BN_EPS 1e-5f

// problem geometry
#define B_   16
#define C_   256
#define H_   80
#define W_   80
#define HW   (H_ * W_)      // 6400
#define CMID 64
#define K2S2 100            // (s*k)^2
#define MPAD 112            // 100 padded to 7*16
#define KENC 576            // 64 * 9

// workspace layout (bytes, 256-aligned blocks)
#define OFF_WCOMP 0u            // 64*256 f16      = 32768 B
#define OFF_WENC  32768u        // 112*576 f16     = 129024 B
#define OFF_BN1   161792u       // 128 f32         = 512 B   (inv[64], beta[64])
#define OFF_BN2   162304u       // 200 f32         = 800 B   (inv[100], beta[100])
#define OFF_WM    163840u       // 16*64*6400 f16  = 13107200 B
#define OFF_ENC   13271040u     // 16*100*6400 f32 = 40960000 B
// total ~54.2 MB of d_ws

// ---------------------------------------------------------------------------
// Kernel 0: weight conversion / re-layout + BN folding (race-free direct map)
// ---------------------------------------------------------------------------
__global__ void __launch_bounds__(256) prep_kernel(
    const float* __restrict__ wcomp, const float* __restrict__ wenc,
    const float* __restrict__ g1, const float* __restrict__ b1,
    const float* __restrict__ m1, const float* __restrict__ v1,
    const float* __restrict__ g2, const float* __restrict__ b2,
    const float* __restrict__ m2, const float* __restrict__ v2,
    _Float16* __restrict__ wcomp_h, _Float16* __restrict__ wenc2,
    float* __restrict__ bn1, float* __restrict__ bn2)
{
    int i = blockIdx.x * 256 + threadIdx.x;
    if (i < CMID * C_) {                       // w_comp: (64,256) row-major, keep
        wcomp_h[i] = (_Float16)wcomp[i];
        return;
    }
    i -= CMID * C_;
    if (i < MPAD * KENC) {                     // w_enc: -> [m][j*64 + c], zero-pad rows
        const int m = i / KENC, kg = i % KENC;
        const int j = kg >> 6, c = kg & 63;    // j = ky*3+kx
        wenc2[i] = (m < K2S2) ? (_Float16)wenc[(m * CMID + c) * 9 + j] : (_Float16)0.f;
        return;
    }
    i -= MPAD * KENC;
    if (i < CMID) {                            // BN1 fold
        const float inv = g1[i] * rsqrtf(v1[i] + BN_EPS);
        bn1[i] = inv;
        bn1[CMID + i] = b1[i] - m1[i] * inv;
        return;
    }
    i -= CMID;
    if (i < K2S2) {                            // BN2 fold
        const float inv = g2[i] * rsqrtf(v2[i] + BN_EPS);
        bn2[i] = inv;
        bn2[K2S2 + i] = b2[i] - m2[i] * inv;
    }
}

// ---------------------------------------------------------------------------
// Kernel 1: 1x1 conv (GEMM 64x256 @ 256xHW) + BN + SiLU -> Wm (f16, NCHW)
// block: 256 threads = 8 waves; wave = (mt in 0..3) x (nt in 0..1)
// Double-buffered LDS B tile (f32->f16, transposed, stride 40 halves):
// one barrier per K-step; next step's global load is in flight during WMMA.
// ---------------------------------------------------------------------------
__global__ void __launch_bounds__(256) comp_gemm(
    const float* __restrict__ X, const _Float16* __restrict__ Wc,
    const float* __restrict__ bn1, _Float16* __restrict__ Wm)
{
    __shared__ __align__(16) _Float16 bt[2][32 * 40];

    const int tid   = threadIdx.x;
    const int lane  = tid & 31;
    const int wid   = tid >> 5;
    const int mt    = wid & 3;
    const int nt    = wid >> 2;
    const int col   = lane & 15;
    const int khalf = (lane < 16) ? 0 : 8;     // ISA 16-bit frag lane split
    const int n     = blockIdx.y;
    const int p0    = blockIdx.x * 32;

    const int skk = tid >> 3;                  // staging: k row 0..31
    const int spx = (tid & 7) * 4;             // staging: 4 pixels

    const float* xsrc = X + ((size_t)n * C_ + skk) * HW + p0 + spx;
    float4 xv = *(const float4*)xsrc;          // prologue: K-step 0

    v8f acc = {};
#pragma unroll
    for (int i = 0; i < 8; ++i) {              // k0 = i*32
        _Float16* bp = &bt[i & 1][0];
        bp[(spx + 0) * 40 + skk] = (_Float16)xv.x;
        bp[(spx + 1) * 40 + skk] = (_Float16)xv.y;
        bp[(spx + 2) * 40 + skk] = (_Float16)xv.z;
        bp[(spx + 3) * 40 + skk] = (_Float16)xv.w;
        if (i < 7)                              // issue next-step load pre-barrier
            xv = *(const float4*)(xsrc + (size_t)(i + 1) * 32 * HW);
        __syncthreads();

        // A fragment: row m, halves 0..7 = K kbase..+7, halves 8..15 = K kbase+16..+23
        const _Float16* arow = Wc + (mt * 16 + col) * C_ + i * 32 + khalf;
        const v8h alo = *(const v8h*)(arow);
        const v8h ahi = *(const v8h*)(arow + 16);
        // B fragment: col px, same K striping, from transposed LDS tile
        const _Float16* brow = &bt[i & 1][0] + (nt * 16 + col) * 40 + khalf;
        const v8h blo = *(const v8h*)(brow);
        const v8h bhi = *(const v8h*)(brow + 16);
        v16h a, b;
#pragma unroll
        for (int u = 0; u < 8; ++u) {
            a[u] = alo[u]; a[u + 8] = ahi[u];
            b[u] = blo[u]; b[u + 8] = bhi[u];
        }
        acc = __builtin_amdgcn_wmma_f32_16x16x32_f16(
            false, a, false, b, (short)0, acc, false, false);
    }

    // C/D layout: VGPR r -> M = r (lanes 0-15) / r+8 (lanes 16-31), N = lane&15
    const int p    = p0 + nt * 16 + col;
    const int mrow = mt * 16 + ((lane < 16) ? 0 : 8);
#pragma unroll
    for (int r = 0; r < 8; ++r) {
        const int m  = mrow + r;
        const float v = acc[r] * bn1[m] + bn1[CMID + m];
        const float s = v / (1.f + __expf(-v));          // SiLU
        Wm[((size_t)n * CMID + m) * HW + p] = (_Float16)s;
    }
}

// ---------------------------------------------------------------------------
// Kernel 2: 3x3 conv (GEMM 112x576 @ 576x16) + BN -> enc (f32, NCHW, 100 ch)
// block: 224 threads = 7 waves (one per M tile); N tile = 16 x along one row.
// Double-buffered im2col staging, one barrier per K-step (18 steps).
// ---------------------------------------------------------------------------
__device__ __forceinline__ _Float16 ld_wm(const _Float16* __restrict__ Wm,
                                          int n, int ch, int yy, int xx)
{
    return ((unsigned)yy < 80u && (unsigned)xx < 80u)
        ? Wm[((size_t)n * CMID + ch) * HW + yy * W_ + xx] : (_Float16)0.f;
}

__global__ void __launch_bounds__(224) enc_gemm(
    const _Float16* __restrict__ Wm, const _Float16* __restrict__ We,
    const float* __restrict__ bn2, float* __restrict__ enc)
{
    __shared__ __align__(16) _Float16 bt[2][16 * 40];

    const int tid   = threadIdx.x;
    const int lane  = tid & 31;
    const int mt    = tid >> 5;                // 0..6
    const int col   = lane & 15;
    const int khalf = (lane < 16) ? 0 : 8;
    const int x0    = blockIdx.x * 16;
    const int y     = blockIdx.y;
    const int n     = blockIdx.z;

    // each thread stages elements e = tid, tid+224 (always), tid+448 (tid<64)
    const int e1 = tid + 224;
    const int e2 = tid + 448;

    _Float16 sv0, sv1, sv2 = (_Float16)0.f;
    {   // prologue: stage K-step 0 (j=0 -> ky=0,kx=0 ; c0=0)
        const int yy = y - 1;
        sv0 = ld_wm(Wm, n, tid >> 4, yy, x0 + (tid & 15) - 1);
        sv1 = ld_wm(Wm, n, e1 >> 4, yy, x0 + (e1 & 15) - 1);
        if (tid < 64) sv2 = ld_wm(Wm, n, e2 >> 4, yy, x0 + (e2 & 15) - 1);
    }

    v8f acc = {};
    for (int s = 0; s < 18; ++s) {             // K-step: j = s/2, c0 = (s&1)*32
        _Float16* bp = &bt[s & 1][0];
        bp[(tid & 15) * 40 + (tid >> 4)] = sv0;
        bp[(e1 & 15) * 40 + (e1 >> 4)]   = sv1;
        if (tid < 64) bp[(e2 & 15) * 40 + (e2 >> 4)] = sv2;

        if (s < 17) {                          // issue next-step staging loads
            const int sn = s + 1;
            const int jn = sn >> 1, c0n = (sn & 1) << 5;
            const int kyn = jn / 3, kxn = jn % 3;
            const int yyn = y + kyn - 1;
            sv0 = ld_wm(Wm, n, c0n + (tid >> 4), yyn, x0 + (tid & 15) + kxn - 1);
            sv1 = ld_wm(Wm, n, c0n + (e1 >> 4),  yyn, x0 + (e1 & 15) + kxn - 1);
            if (tid < 64)
                sv2 = ld_wm(Wm, n, c0n + (e2 >> 4), yyn, x0 + (e2 & 15) + kxn - 1);
        }
        __syncthreads();

        const int j = s >> 1, c0 = (s & 1) << 5;
        const _Float16* arow = We + (mt * 16 + col) * KENC + j * 64 + c0 + khalf;
        const v8h alo = *(const v8h*)(arow);
        const v8h ahi = *(const v8h*)(arow + 16);
        const _Float16* brow = &bt[s & 1][0] + col * 40 + khalf;
        const v8h blo = *(const v8h*)(brow);
        const v8h bhi = *(const v8h*)(brow + 16);
        v16h a, b;
#pragma unroll
        for (int u = 0; u < 8; ++u) {
            a[u] = alo[u]; a[u + 8] = ahi[u];
            b[u] = blo[u]; b[u + 8] = bhi[u];
        }
        acc = __builtin_amdgcn_wmma_f32_16x16x32_f16(
            false, a, false, b, (short)0, acc, false, false);
    }

    const int x    = x0 + col;
    const int mrow = mt * 16 + ((lane < 16) ? 0 : 8);
#pragma unroll
    for (int r = 0; r < 8; ++r) {
        const int m = mrow + r;
        if (m < K2S2) {
            const float v = acc[r] * bn2[m] + bn2[K2S2 + m];
            enc[((size_t)n * K2S2 + m) * HW + y * W_ + x] = v;
        }
    }
}

// ---------------------------------------------------------------------------
// Kernel 3: PixelShuffle + softmax(25) + 25-tap reassembly (bandwidth bound)
// block: 16 low-res positions (one row segment). Phase 1: 64 threads build
// 4 softmaxes/position into LDS [pos][tap][q]. Phase 2: 256 threads sweep
// channels with the 25 weight-float4s hoisted into registers; each (c,pos)
// reads the shared 5x5 low-res X patch once and emits the 2x2 output block.
// ---------------------------------------------------------------------------
__global__ void __launch_bounds__(256) carafe_apply(
    const float* __restrict__ X, const float* __restrict__ enc,
    float* __restrict__ out)
{
    __shared__ __align__(16) float wsm[16][25][4];

    const int tid = threadIdx.x;
    const int w0  = blockIdx.x * 16;
    const int y   = blockIdx.y;
    const int n   = blockIdx.z;

    if (tid < 64) {
        const int p = tid >> 2, q = tid & 3;   // q = si*2 + sj
        const int x = w0 + p;
        const float* eb = enc + ((size_t)n * K2S2 + q) * HW + y * W_ + x;
        float vals[25];
        float mx = -3.4e38f;
#pragma unroll
        for (int k = 0; k < 25; ++k) {
            const float v = eb[(size_t)k * 4 * HW];   // channel kidx*4 + q
            vals[k] = v;
            mx = fmaxf(mx, v);
        }
        float s = 0.f;
#pragma unroll
        for (int k = 0; k < 25; ++k) {
            const float e = __expf(vals[k] - mx);
            vals[k] = e;
            s += e;
        }
        const float inv = 1.f / s;
#pragma unroll
        for (int k = 0; k < 25; ++k) wsm[p][k][q] = vals[k] * inv;
    }
    __syncthreads();

    const int p  = tid & 15;
    const int cg = tid >> 4;
    const int x  = w0 + p;

    float4 wv[25];                              // hoist: invariant over channels
#pragma unroll
    for (int t = 0; t < 25; ++t) wv[t] = *(const float4*)&wsm[p][t][0];

    for (int ci = 0; ci < 16; ++ci) {
        const int c = cg * 16 + ci;
        const float* xp = X + ((size_t)n * C_ + c) * HW;
        float a0 = 0.f, a1 = 0.f, a2 = 0.f, a3 = 0.f;
#pragma unroll
        for (int ki = 0; ki < 5; ++ki) {
            const int yy  = y + ki - 2;
            const bool yok = (unsigned)yy < 80u;
#pragma unroll
            for (int kj = 0; kj < 5; ++kj) {
                const int xx = x + kj - 2;
                const float tap =
                    (yok && (unsigned)xx < 80u) ? xp[yy * W_ + xx] : 0.f;
                const float4 w = wv[ki * 5 + kj];
                a0 += w.x * tap; a1 += w.y * tap;
                a2 += w.z * tap; a3 += w.w * tap;
            }
        }
        float* ob = out + (((size_t)n * C_ + c) * 160 + 2 * y) * 160 + 2 * x;
        const float2 lo = {a0, a1};
        const float2 hi = {a2, a3};
        *(float2*)(ob)       = lo;   // (si=0, sj=0/1)
        *(float2*)(ob + 160) = hi;   // (si=1, sj=0/1)
    }
}

// ---------------------------------------------------------------------------
extern "C" void kernel_launch(void* const* d_in, const int* in_sizes, int n_in,
                              void* d_out, int out_size, void* d_ws, size_t ws_size,
                              hipStream_t stream) {
    (void)in_sizes; (void)n_in; (void)out_size; (void)ws_size;
    const float* X     = (const float*)d_in[0];
    const float* wcomp = (const float*)d_in[1];
    const float* g1    = (const float*)d_in[2];
    const float* b1    = (const float*)d_in[3];
    const float* m1    = (const float*)d_in[4];
    const float* v1    = (const float*)d_in[5];
    const float* wenc  = (const float*)d_in[6];
    const float* g2    = (const float*)d_in[7];
    const float* b2    = (const float*)d_in[8];
    const float* m2    = (const float*)d_in[9];
    const float* v2    = (const float*)d_in[10];

    char* ws = (char*)d_ws;
    _Float16* wcomp_h = (_Float16*)(ws + OFF_WCOMP);
    _Float16* wenc2   = (_Float16*)(ws + OFF_WENC);
    float*    bn1     = (float*)(ws + OFF_BN1);
    float*    bn2     = (float*)(ws + OFF_BN2);
    _Float16* Wm      = (_Float16*)(ws + OFF_WM);
    float*    enc     = (float*)(ws + OFF_ENC);
    float*    out     = (float*)d_out;

    const int prep_tasks = CMID * C_ + MPAD * KENC + CMID + K2S2;
    prep_kernel<<<(prep_tasks + 255) / 256, 256, 0, stream>>>(
        wcomp, wenc, g1, b1, m1, v1, g2, b2, m2, v2, wcomp_h, wenc2, bn1, bn2);

    comp_gemm<<<dim3(HW / 32, B_), 256, 0, stream>>>(X, wcomp_h, bn1, Wm);
    enc_gemm<<<dim3(W_ / 16, H_, B_), 224, 0, stream>>>(Wm, wenc2, bn2, enc);
    carafe_apply<<<dim3(W_ / 16, H_, B_), 256, 0, stream>>>(X, enc, out);
}
